// EGATConv_A1_83330955477491
// MI455X (gfx1250) — matrix operand order repaired and checked by reading, not compiled, not run on hardware
//
#include <hip/hip_runtime.h>
#include <hip/hip_bf16.h>
#include <stdint.h>

// ---------------------------------------------------------------------------
// EGATConv forward for MI455X (gfx1250, wave32, WMMA).
// GEMMs in bf16 WMMA 16x16x32 / f32 accumulate. Edge gather staged with
// GLOBAL_LOAD_ASYNC_TO_LDS (ASYNCcnt). Softmax/scatter are HBM/atomic bound.
// ---------------------------------------------------------------------------

#define NN 50000          // nodes
#define NE 1000000        // edges
#define IN_N 128
#define IN_E 64
#define OUT_N 128
#define OUT_E 64
#define NH 4
#define KSTACK (IN_E + 2*IN_N)   // 320
#define NCOLS_E (OUT_E*NH)       // 256
#define NCOLS_N (OUT_N*NH)       // 512
#define EDGES_PER_BLK 64         // 4 row-tiles of 16

typedef __bf16 bf16_t;
typedef __attribute__((ext_vector_type(16))) __bf16 v16bf;
typedef __attribute__((ext_vector_type(8)))  __bf16 v8bf;
typedef __attribute__((ext_vector_type(8)))  float  v8f;

// ---------------- WMMA helpers --------------------------------------------

__device__ inline v8f wmma_bf16(v16bf a, v16bf b, v8f c) {
    // (neg_a, A, neg_b, B, c_mod, C, reuse_a, reuse_b)
    return __builtin_amdgcn_wmma_f32_16x16x32_bf16(
        false, a, false, b, (short)0, c, false, false);
}

// A-fragment from a row-major LDS tile row (16-bit A 16x32 layout, ISA 7.12.2):
// lanes 0-15 hold M=lane, K={k0..k0+7, k0+16..k0+23};
// lanes 16-31 hold M=lane-16, K={k0+8..k0+15, k0+24..k0+31}.
__device__ inline v16bf load_a_frag(const bf16_t* row, int k0, int lane) {
    int off = (lane & 16) ? 8 : 0;
    v8bf lo = *(const v8bf*)(row + k0 + off);
    v8bf hi = *(const v8bf*)(row + k0 + 16 + off);
    v16bf a;
#pragma unroll
    for (int j = 0; j < 8; ++j) { a[j] = lo[j]; a[j + 8] = hi[j]; }
    return a;
}

// ---------------- Weight pre-pack into B-fragment layout -------------------
// B 32x16 (16-bit): lanes 0-15 hold N=lane, K=kt*32+{0..7, 16..23};
//                   lanes 16-31 hold N=lane-16, K=kt*32+{8..15, 24..31}.
// Packed so each lane's 16-bf16 fragment is one contiguous 32B load:
//   frag[( (kt*NT + nt)*32 + lane )*16 + j]
__global__ __launch_bounds__(128) void pack_b_frags(const float* __restrict__ W,
                                                    bf16_t* __restrict__ out,
                                                    int K, int N) {
    int tile = blockIdx.x * 4 + (threadIdx.x >> 5);
    int lane = threadIdx.x & 31;
    int KT = K >> 5, NT = N >> 4;
    if (tile >= KT * NT) return;
    int kt = tile / NT, nt = tile % NT;
    int n  = nt * 16 + (lane & 15);
    int kb = kt * 32 + ((lane & 16) ? 8 : 0);
    bf16_t* d = out + ((size_t)tile * 32 + lane) * 16;
#pragma unroll
    for (int j = 0; j < 16; ++j)
        d[j] = (bf16_t)W[(size_t)(kb + (j & 7) + ((j & 8) << 1)) * N + n];
}

// ---------------- init ------------------------------------------------------
__global__ void init_kernel(float* h_out, unsigned* nmax, float* denom) {
    size_t i = (size_t)blockIdx.x * blockDim.x + threadIdx.x;
    if (i < (size_t)NN * NCOLS_N) h_out[i] = 0.0f;
    if (i < (size_t)NN * NH) { nmax[i] = 0u; denom[i] = 0.0f; }
}

// ---------------- node projection: h = nfeats @ W_n + b_n ------------------
__global__ __launch_bounds__(128) void node_proj_kernel(
        const float* __restrict__ nfeats, const bf16_t* __restrict__ Wn_frag,
        const float* __restrict__ b_n, float* __restrict__ h_f32) {
    __shared__ bf16_t As[16 * IN_N];
    int tid = threadIdx.x;
    int node0 = blockIdx.x * 16;
    for (int i = tid; i < 16 * IN_N; i += 128)
        As[i] = (bf16_t)nfeats[(size_t)node0 * IN_N + i];
    __syncthreads();
    int wave = tid >> 5, lane = tid & 31;
    const bf16_t* arow = As + (lane & 15) * IN_N;
    for (int nt = wave; nt < NCOLS_N / 16; nt += 4) {   // 32 col-tiles
        v8f c = {};
#pragma unroll
        for (int kt = 0; kt < IN_N / 32; ++kt) {        // 4 k-steps
            v16bf a = load_a_frag(arow, kt * 32, lane);
            v16bf b = *(const v16bf*)(Wn_frag +
                (((size_t)kt * (NCOLS_N / 16) + nt) * 32 + lane) * 16);
            c = wmma_bf16(a, b, c);
        }
        int col = nt * 16 + (lane & 15);
        float bias = b_n[col];
        int rbase = (lane & 16) ? 8 : 0;
#pragma unroll
        for (int r = 0; r < 8; ++r)
            h_f32[(size_t)(node0 + rbase + r) * NCOLS_N + col] = c[r] + bias;
    }
}

// ---------------- head mean -> bf16 ----------------------------------------
__global__ void mean_kernel(const float* __restrict__ h_f32,
                            bf16_t* __restrict__ hmean) {
    size_t i = (size_t)blockIdx.x * blockDim.x + threadIdx.x;
    if (i >= (size_t)NN * OUT_N) return;
    size_t n = i >> 7; int c = (int)(i & 127);
    const float* p = h_f32 + n * NCOLS_N + c;
    hmean[i] = (bf16_t)((p[0] + p[128] + p[256] + p[384]) * 0.25f);
}

// ---------------- edge projection: ef = bf16(efeats @ W_ep + b_ep) ---------
__global__ __launch_bounds__(128) void edge_proj_kernel(
        const float* __restrict__ efeats, const bf16_t* __restrict__ Wep_frag,
        const float* __restrict__ b_ep, bf16_t* __restrict__ ef_bf16) {
    __shared__ bf16_t As[16 * IN_E];
    int tid = threadIdx.x;
    long e0 = (long)blockIdx.x * 16;
    for (int i = tid; i < 16 * IN_E; i += 128)
        As[i] = (bf16_t)efeats[e0 * IN_E + i];
    __syncthreads();
    int wave = tid >> 5, lane = tid & 31;
    const bf16_t* arow = As + (lane & 15) * IN_E;
    int nt = wave;                                    // 4 col-tiles, 1 per wave
    v8f c = {};
#pragma unroll
    for (int kt = 0; kt < IN_E / 32; ++kt) {
        v16bf a = load_a_frag(arow, kt * 32, lane);
        v16bf b = *(const v16bf*)(Wep_frag +
            (((size_t)kt * 4 + nt) * 32 + lane) * 16);
        c = wmma_bf16(a, b, c);
    }
    int col = nt * 16 + (lane & 15);
    float bias = b_ep[col];
    int rbase = (lane & 16) ? 8 : 0;
#pragma unroll
    for (int r = 0; r < 8; ++r)
        ef_bf16[(e0 + rbase + r) * IN_E + col] = (bf16_t)(c[r] + bias);
}

// ---------------- main edge GEMM: f_out = leakyrelu(stack @ W_e) -----------
// 64 edges per block (4 row-tiles of 16). A-stack staged to LDS with
// GLOBAL_LOAD_ASYNC_TO_LDS (ASYNCcnt); each B fragment is reused across the
// 4 row-tiles (4 accumulators per wave).
__global__ __launch_bounds__(128) void edge_attn_gemm_kernel(
        const bf16_t* __restrict__ hmean, const bf16_t* __restrict__ ef_bf16,
        const int* __restrict__ src, const int* __restrict__ dst,
        const bf16_t* __restrict__ We_frag, float* __restrict__ f_out) {
    __shared__ bf16_t As[EDGES_PER_BLK * KSTACK];      // 40 KB
    int tid = threadIdx.x;
    long e0 = (long)blockIdx.x * EDGES_PER_BLK;
    int wave = tid >> 5, lane = tid & 31;

    // ---- async stage: wave w stages edges w*16 .. w*16+15 ----
    // LDS row layout (bytes): [src hmean 0..255 | ef 256..383 | dst hmean 384..639]
    unsigned ldsbase = (unsigned)(uintptr_t)(&As[0]);
    for (int er = 0; er < 16; ++er) {
        int el = wave * 16 + er;
        long e = e0 + el;
        const char* srow = (const char*)(hmean + (size_t)src[e] * OUT_N);
        const char* drow = (const char*)(hmean + (size_t)dst[e] * OUT_N);
        const char* erow = (const char*)(ef_bf16 + e * IN_E);
        unsigned lrow = ldsbase + (unsigned)el * (KSTACK * 2);
        // 32 lanes x 16B: lanes 0-15 -> src half, lanes 16-31 -> dst half
        const void* g128 = (lane < 16) ? (const void*)(srow + lane * 16)
                                       : (const void*)(drow + (lane - 16) * 16);
        unsigned l128 = (lane < 16) ? (lrow + lane * 16)
                                    : (lrow + 384 + (lane - 16) * 16);
        asm volatile("global_load_async_to_lds_b128 %0, %1, off"
                     :: "v"(l128), "v"(g128) : "memory");
        // 32 lanes x 4B: ef row (128 B)
        const void* g32 = (const void*)(erow + lane * 4);
        unsigned l32 = lrow + 256 + lane * 4;
        asm volatile("global_load_async_to_lds_b32 %0, %1, off"
                     :: "v"(l32), "v"(g32) : "memory");
    }
    asm volatile("s_wait_asynccnt 0" ::: "memory");
    __syncthreads();

    // ---- compute: wave w owns col-tiles w*4..w*4+3; 4 row-tiles each ----
    const bf16_t* arow = As + (lane & 15) * KSTACK;    // row m of row-tile 0
    int col0 = (lane & 15);
    int rbase = (lane & 16) ? 8 : 0;
#pragma unroll
    for (int t = 0; t < 4; ++t) {
        int nt = wave * 4 + t;                         // 16 col-tiles total
        v8f c0 = {}, c1 = {}, c2 = {}, c3 = {};
        for (int kt = 0; kt < KSTACK / 32; ++kt) {     // 10 k-steps
            const bf16_t* bp = We_frag +
                (((size_t)kt * (NCOLS_E / 16) + nt) * 32 + lane) * 16;
            if (kt + 1 < KSTACK / 32)
                __builtin_prefetch(bp + (NCOLS_E / 16) * 32 * 16, 0, 1);
            v16bf b = *(const v16bf*)bp;
            c0 = wmma_bf16(load_a_frag(arow + 0 * 16 * KSTACK, kt * 32, lane), b, c0);
            c1 = wmma_bf16(load_a_frag(arow + 1 * 16 * KSTACK, kt * 32, lane), b, c1);
            c2 = wmma_bf16(load_a_frag(arow + 2 * 16 * KSTACK, kt * 32, lane), b, c2);
            c3 = wmma_bf16(load_a_frag(arow + 3 * 16 * KSTACK, kt * 32, lane), b, c3);
        }
        int col = nt * 16 + col0;
        v8f cc[4] = {c0, c1, c2, c3};
#pragma unroll
        for (int rt = 0; rt < 4; ++rt) {
            long erow0 = e0 + rt * 16 + rbase;
#pragma unroll
            for (int r = 0; r < 8; ++r) {
                float v = cc[rt][r];
                v = v > 0.0f ? v : 0.01f * v;          // leaky_relu(0.01)
                f_out[(erow0 + r) * (size_t)NCOLS_E + col] = v;
            }
        }
    }
}

// ---------------- attention logits + segment max ---------------------------
__device__ inline unsigned ford(float x) {
    unsigned u = __float_as_uint(x);
    return (u & 0x80000000u) ? ~u : (u | 0x80000000u);
}
__device__ inline float funord(unsigned u) {
    return (u & 0x80000000u) ? __uint_as_float(u & 0x7fffffffu)
                             : __uint_as_float(~u);
}

__global__ void attn_logit_kernel(const float* __restrict__ f_out,
                                  const float* __restrict__ W_a,
                                  const int* __restrict__ dst,
                                  float* __restrict__ a_buf,
                                  unsigned* __restrict__ nmax) {
    long i = (long)blockIdx.x * blockDim.x + threadIdx.x;
    if (i >= (long)NE * NH) return;
    long e = i >> 2; int h = (int)(i & 3);
    const float* f = f_out + e * (size_t)NCOLS_E + h * OUT_E;
    float s = 0.0f;
#pragma unroll
    for (int c = 0; c < OUT_E; ++c) s += f[c] * W_a[c];
    a_buf[i] = s;
    atomicMax(&nmax[(size_t)dst[e] * NH + h], ford(s));
}

// ---------------- exp + segment sum ----------------------------------------
__global__ void softmax_num_kernel(const int* __restrict__ dst,
                                   const unsigned* __restrict__ nmax,
                                   float* __restrict__ a_buf,
                                   float* __restrict__ denom) {
    long i = (long)blockIdx.x * blockDim.x + threadIdx.x;
    if (i >= (long)NE * NH) return;
    long e = i >> 2; int h = (int)(i & 3);
    float m  = funord(nmax[(size_t)dst[e] * NH + h]);
    float ex = __expf(a_buf[i] - m);
    a_buf[i] = ex;
    atomicAdd(&denom[(size_t)dst[e] * NH + h], ex);
}

// ---------------- alpha-weighted scatter-sum of h[src] ---------------------
__global__ __launch_bounds__(128) void scatter_kernel(
        const int* __restrict__ src, const int* __restrict__ dst,
        const float* __restrict__ a_buf, const float* __restrict__ denom,
        const float* __restrict__ h_f32, float* __restrict__ h_out) {
    long e = blockIdx.x;
    int t = threadIdx.x;
    int s = src[e], d = dst[e];
#pragma unroll
    for (int j = 0; j < 4; ++j) {
        int col = t + j * 128;
        int h = col >> 7;
        float alpha = a_buf[e * NH + h] / denom[(size_t)d * NH + h];
        atomicAdd(&h_out[(size_t)d * NCOLS_N + col],
                  alpha * h_f32[(size_t)s * NCOLS_N + col]);
    }
}

// ---------------------------------------------------------------------------
extern "C" void kernel_launch(void* const* d_in, const int* in_sizes, int n_in,
                              void* d_out, int out_size, void* d_ws, size_t ws_size,
                              hipStream_t stream) {
    (void)in_sizes; (void)n_in; (void)out_size; (void)ws_size;
    const float* nfeats = (const float*)d_in[0];
    const float* efeats = (const float*)d_in[1];
    const int*   src    = (const int*)d_in[2];
    const int*   dst    = (const int*)d_in[3];
    /* d_in[4] edge_type: unused by the reference math */
    const float* W_ep   = (const float*)d_in[5];
    const float* b_ep   = (const float*)d_in[6];
    const float* W_n    = (const float*)d_in[7];
    const float* b_n    = (const float*)d_in[8];
    const float* W_e    = (const float*)d_in[9];
    const float* W_a    = (const float*)d_in[10];

    // workspace layout (256B-aligned regions)
    char* ws = (char*)d_ws;
    float*    h_f32    = (float*)   (ws + 0);                    // 102,400,000 B
    bf16_t*   hmean    = (bf16_t*)  (ws + 102400000);            //  12,800,000 B
    bf16_t*   ef_bf16  = (bf16_t*)  (ws + 115200000);            // 128,000,000 B
    float*    a_buf    = (float*)   (ws + 243200000);            //  16,000,000 B
    unsigned* nmax     = (unsigned*)(ws + 259200000);            //     800,000 B
    float*    denom    = (float*)   (ws + 260000000);            //     800,000 B
    bf16_t*   Wn_frag  = (bf16_t*)  (ws + 260800000);            //     131,072 B
    bf16_t*   Wep_frag = (bf16_t*)  (ws + 260931328);            //       8,192 B
    bf16_t*   We_frag  = (bf16_t*)  (ws + 260939776);            //     163,840 B

    float* h_out = (float*)d_out;                                // [NN, NH*OUT_N]
    float* f_out = (float*)d_out + (size_t)NN * NCOLS_N;         // [NE, NH*OUT_E]

    // 1) pack weights into WMMA B-fragment layout (tiny, L2-resident)
    pack_b_frags<<<dim3((128 + 3) / 4), 128, 0, stream>>>(W_n,  Wn_frag,  IN_N,   NCOLS_N); // 128 tiles
    pack_b_frags<<<dim3((8   + 3) / 4), 128, 0, stream>>>(W_ep, Wep_frag, IN_E,   IN_E);    //   8 tiles
    pack_b_frags<<<dim3((160 + 3) / 4), 128, 0, stream>>>(W_e,  We_frag,  KSTACK, NCOLS_E); // 160 tiles

    // 2) zero accumulators
    init_kernel<<<dim3(((size_t)NN * NCOLS_N + 255) / 256), 256, 0, stream>>>(h_out, nmax, denom);

    // 3) node projection GEMM + head mean
    node_proj_kernel<<<dim3(NN / 16), 128, 0, stream>>>(nfeats, Wn_frag, b_n, h_f32);
    mean_kernel<<<dim3(((size_t)NN * OUT_N + 255) / 256), 256, 0, stream>>>(h_f32, hmean);

    // 4) edge projection GEMM
    edge_proj_kernel<<<dim3(NE / 16), 128, 0, stream>>>(efeats, Wep_frag, b_ep, ef_bf16);

    // 5) dominant edge GEMM (async gather + WMMA + leaky_relu) -> f_out
    edge_attn_gemm_kernel<<<dim3(NE / EDGES_PER_BLK), 128, 0, stream>>>(
        hmean, ef_bf16, src, dst, We_frag, f_out);

    // 6) logits + segment softmax
    attn_logit_kernel<<<dim3(((long)NE * NH + 255) / 256), 256, 0, stream>>>(
        f_out, W_a, dst, a_buf, nmax);
    softmax_num_kernel<<<dim3(((long)NE * NH + 255) / 256), 256, 0, stream>>>(
        dst, nmax, a_buf, denom);

    // 7) weighted scatter-sum -> h_out
    scatter_kernel<<<dim3(NE), 128, 0, stream>>>(src, dst, a_buf, denom, h_f32, h_out);
}
